// hardwiredAttention_66374424593210
// MI455X (gfx1250) — matrix operand order repaired and checked by reading, compiled, not run on hardware
//
#include <hip/hip_runtime.h>

#define Bb 64
#define Nn 512
#define Hh 128
#define NCOG 72
#define NRB 72
#define ASTR 40   // padded LDS stride (bf16 elems) for A slab rows (80B, 16B-aligned)
#define BSTR 40   // padded LDS stride (bf16 elems) for B slab rows

typedef __attribute__((ext_vector_type(16))) __bf16 v16bf;
typedef __attribute__((ext_vector_type(8)))  float  v8f;

union FragU {
    v16bf v;
    uint4 q[2];
};

static __device__ __forceinline__ unsigned short f2bf(float f) {
    unsigned u = __float_as_uint(f);
    unsigned r = u + 0x7FFFu + ((u >> 16) & 1u);   // round-to-nearest-even
    return (unsigned short)(r >> 16);
}

// Async 16-byte copy global -> LDS (CDNA5 GLOBAL_LOAD_ASYNC_TO_LDS_B128, ASYNCcnt)
static __device__ __forceinline__ void async_copy_b128(void* lds_dst, const void* gsrc) {
    unsigned lds_addr = (unsigned)(size_t)lds_dst;            // low 32 bits = LDS offset
    unsigned long long ga = (unsigned long long)(size_t)gsrc;
    asm volatile("global_load_async_to_lds_b128 %0, %1, off"
                 :: "v"(lds_addr), "v"(ga)
                 : "memory");
}

static __device__ __forceinline__ void wait_async_zero() {
    asm volatile("s_wait_asynccnt 0" ::: "memory");
}

// ---------------------------------------------------------------------------
// Kernel 1: transpose h_t [N,B,H] f32 -> hbT [B,H,N] bf16 (LDS 32x32 tiles),
//           and zero the global-max cell.
// ---------------------------------------------------------------------------
__global__ void hw_transpose_h(const float* __restrict__ h,
                               unsigned short* __restrict__ hbT,
                               unsigned* __restrict__ maxp) {
    __shared__ unsigned short tile[32 * 36];   // [j_local][h_local], padded

    const int tid = threadIdx.x;
    if (blockIdx.x == 0 && blockIdx.y == 0 && blockIdx.z == 0 && tid == 0)
        *maxp = 0u;

    const int j0 = blockIdx.x * 32;
    const int h0 = blockIdx.y * 32;
    const int b  = blockIdx.z;

    // read: coalesced along h
    {
        int jl = tid >> 3;
        int he = (tid & 7) * 4;
        float4 v = *(const float4*)(h + (size_t)(j0 + jl) * Bb * Hh
                                      + (size_t)b * Hh + h0 + he);
        uint2 packed;
        packed.x = (unsigned)f2bf(v.x) | ((unsigned)f2bf(v.y) << 16);
        packed.y = (unsigned)f2bf(v.z) | ((unsigned)f2bf(v.w) << 16);
        *(uint2*)(&tile[jl * 36 + he]) = packed;
    }
    __syncthreads();
    // write: coalesced along j
    {
        int hl = tid >> 3;
        int je = (tid & 7) * 4;
        unsigned short s0 = tile[(je + 0) * 36 + hl];
        unsigned short s1 = tile[(je + 1) * 36 + hl];
        unsigned short s2 = tile[(je + 2) * 36 + hl];
        unsigned short s3 = tile[(je + 3) * 36 + hl];
        uint2 packed;
        packed.x = (unsigned)s0 | ((unsigned)s1 << 16);
        packed.y = (unsigned)s2 | ((unsigned)s3 << 16);
        *(uint2*)(hbT + (size_t)b * Hh * Nn + (size_t)(h0 + hl) * Nn + j0 + je) = packed;
    }
}

// ---------------------------------------------------------------------------
// Kernel 2: w_un[b,i,j] = mask[b,i]*mask[b,j]*relu(domain[cog,r] - d) -> bf16,
//           plus global max via wave-shuffle + block LDS + one atomic/block.
// ---------------------------------------------------------------------------
__global__ void hw_pass1(const int*   __restrict__ rmat,
                         const float* __restrict__ dmat,
                         const float* __restrict__ mask,
                         const int*   __restrict__ cogmat,
                         const float* __restrict__ domain,
                         unsigned short* __restrict__ wbf,
                         unsigned* __restrict__ maxp) {
    __shared__ float sdom[NCOG * NRB];
    __shared__ float smax[8];

    const int tid = threadIdx.x;
    for (int t = tid; t < NCOG * NRB; t += 256) sdom[t] = domain[t];
    __syncthreads();

    size_t base = ((size_t)blockIdx.x * 256 + tid) * 4;
    int b   = (int)(base / ((size_t)Nn * Nn));
    int rem = (int)(base % ((size_t)Nn * Nn));
    int i = rem / Nn;
    int j = rem % Nn;

    int4   cg = *(const int4*)(cogmat + base);
    int4   rr = *(const int4*)(rmat + base);
    float4 dd = *(const float4*)(dmat + base);
    float  mi = mask[b * Nn + i];
    float4 mj = *(const float4*)(mask + b * Nn + j);

    float w0 = mi * mj.x * fmaxf(sdom[cg.x * NRB + rr.x] - dd.x, 0.0f);
    float w1 = mi * mj.y * fmaxf(sdom[cg.y * NRB + rr.y] - dd.y, 0.0f);
    float w2 = mi * mj.z * fmaxf(sdom[cg.z * NRB + rr.z] - dd.z, 0.0f);
    float w3 = mi * mj.w * fmaxf(sdom[cg.w * NRB + rr.w] - dd.w, 0.0f);

    uint2 packed;
    packed.x = (unsigned)f2bf(w0) | ((unsigned)f2bf(w1) << 16);
    packed.y = (unsigned)f2bf(w2) | ((unsigned)f2bf(w3) << 16);
    *(uint2*)(wbf + base) = packed;

    // local -> wave32 -> block max reduction, then one atomic per block
    float lm = fmaxf(fmaxf(w0, w1), fmaxf(w2, w3));
    #pragma unroll
    for (int off = 16; off > 0; off >>= 1)
        lm = fmaxf(lm, __shfl_xor(lm, off, 32));
    if ((tid & 31) == 0) smax[tid >> 5] = lm;
    __syncthreads();
    if (tid == 0) {
        float m = smax[0];
        #pragma unroll
        for (int k = 1; k < 8; ++k) m = fmaxf(m, smax[k]);
        atomicMax(maxp, __float_as_uint(m));   // valid: all values >= 0
    }
}

// ---------------------------------------------------------------------------
// Kernel 3: out[b,i,h] = (1/max) * sum_j w_un[b,i,j] * hbT[b,h,j]
// 256 threads = 8 waves; each block does a 128(i) x 16(h) slab of batch b.
// Async global->LDS staging (ASYNCcnt), v_wmma_f32_16x16x32_bf16, 16 K-steps.
// ---------------------------------------------------------------------------
__global__ void hw_gemm(const unsigned short* __restrict__ wbf,
                        const unsigned short* __restrict__ hbT,
                        const unsigned* __restrict__ maxp,
                        float* __restrict__ out) {
    __shared__ unsigned short aslab[128 * ASTR];  // [i_local][k] padded
    __shared__ unsigned short bslab[16 * BSTR];   // [n(h)][k]   padded

    const int tid  = threadIdx.x;
    const int wave = tid >> 5;
    const int lane = tid & 31;
    const int half = lane >> 4;
    const int n    = lane & 15;

    const int h0 = blockIdx.x * 16;
    const int i0 = blockIdx.y * 128;
    const int b  = blockIdx.z;

    const float invmax = 1.0f / __uint_as_float(*maxp);

    v8f acc = {};

    // per-thread staging assignments (fixed across k-steps)
    const int a_il = tid >> 1;
    const int a_c  = (tid & 1) * 16;
    const unsigned short* a_src0 =
        wbf + (size_t)b * Nn * Nn + (size_t)(i0 + a_il) * Nn + a_c;
    unsigned short* a_dst = &aslab[a_il * ASTR + a_c];

    const int b_n = tid >> 1;           // valid for tid < 32
    const int b_c = (tid & 1) * 16;
    const unsigned short* b_src0 =
        hbT + (size_t)b * Hh * Nn + (size_t)(h0 + b_n) * Nn + b_c;
    unsigned short* b_dst = &bslab[b_n * BSTR + b_c];

    for (int j0 = 0; j0 < Nn; j0 += 32) {
        __syncthreads();                     // previous tile fully consumed
        // ---- async stage A: 256 threads x 16B cover k 0..7,16..23 of the 128x32 slab ----
        async_copy_b128(a_dst, a_src0 + j0);
        // ---- async stage B: wave 0 covers k 0..7,16..23 of the 16x32 slab (contiguous rows of hbT) ----
        if (tid < 32)
            async_copy_b128(b_dst, b_src0 + j0);
        wait_async_zero();                   // this wave's transfers landed
        __syncthreads();                     // all waves' transfers landed

        // ---- fragments per CDNA5 16-bit layouts ----
        // A 16x32: lane M = lane%16, halves select K blocks {0..7,16..23}/{8..15,24..31}
        FragU fa, fb;
        const int arow = (wave * 16 + n) * ASTR;
        fa.q[0] = *(const uint4*)(&aslab[arow + half * 8]);
        fa.q[1] = *(const uint4*)(&aslab[arow + 16 + half * 8]);
        // B 32x16: lane N = lane%16, halves select K blocks {0..7,16..23}/{8..15,24..31}
        const int brow = n * BSTR + half * 8;
        fb.q[0] = *(const uint4*)(&bslab[brow]);
        fb.q[1] = *(const uint4*)(&bslab[brow + 16]);

        acc = __builtin_amdgcn_wmma_f32_16x16x32_bf16(
            false, fa.v, false, fb.v, (short)0, acc, false, false);
    }

    // ---- C/D layout: VGPR r -> row wave*16 + half*8 + r, col n ----
    float* op = out + (size_t)b * Nn * Hh
                    + (size_t)(i0 + wave * 16 + half * 8) * Hh
                    + h0 + n;
    #pragma unroll
    for (int r = 0; r < 8; ++r)
        op[(size_t)r * Hh] = acc[r] * invmax;
}

// ---------------------------------------------------------------------------
extern "C" void kernel_launch(void* const* d_in, const int* in_sizes, int n_in,
                              void* d_out, int out_size, void* d_ws, size_t ws_size,
                              hipStream_t stream) {
    const float* h_t    = (const float*)d_in[0];   // [N,B,H]
    const int*   r_mat  = (const int*)  d_in[1];   // [B,N,N]
    const float* d_mat  = (const float*)d_in[2];   // [B,N,N]
    const float* mask   = (const float*)d_in[3];   // [B,N]
    const int*   cogmat = (const int*)  d_in[4];   // [B,N,N]
    const float* domain = (const float*)d_in[5];   // [72,72]
    float* out = (float*)d_out;                    // [B,N,H]

    char* ws = (char*)d_ws;
    unsigned*       maxp = (unsigned*)ws;
    unsigned short* hbT  = (unsigned short*)(ws + 256);
    unsigned short* wbf  = (unsigned short*)(ws + 256 + (size_t)Bb * Hh * Nn * 2);

    // 1) h_t -> bf16, transposed to [B,H,N] (+ zero global max cell)
    {
        dim3 grid(Nn / 32, Hh / 32, Bb);           // (16, 4, 64)
        hw_transpose_h<<<grid, 256, 0, stream>>>(h_t, hbT, maxp);
    }
    // 2) masked/relu/gather weights -> bf16 + global max
    {
        size_t total = (size_t)Bb * Nn * Nn;       // 16,777,216
        hw_pass1<<<(unsigned)(total / (256 * 4)), 256, 0, stream>>>(
            r_mat, d_mat, mask, cogmat, domain, wbf, maxp);
    }
    // 3) WMMA GEMM with folded 1/max, async LDS staging
    {
        dim3 grid(Hh / 16, Nn / 128, Bb);          // (8, 4, 64)
        hw_gemm<<<grid, 256, 0, stream>>>(wbf, hbT, maxp, out);
    }
}
